// CompoundProteinInteractionPrediction_55104430407749
// MI455X (gfx1250) — matrix operand order repaired and checked by reading, not compiled, not run on hardware
//
#include <hip/hip_runtime.h>
#include <hip/hip_bf16.h>

// ---------------------------------------------------------------------------
// Types for CDNA5 WMMA (wave32): v_wmma_f32_16x16x32_bf16
// ---------------------------------------------------------------------------
typedef __attribute__((ext_vector_type(16))) __bf16 v16bf;
typedef __attribute__((ext_vector_type(8)))  float  v8f;

union FragBF {
    uint4 q[2];   // two 16-byte halves
    v16bf v;      // 16 bf16 lanes-worth
};

__device__ __forceinline__ unsigned short f2bf(float f) {
    unsigned int u = __float_as_uint(f);
    unsigned int r = u + 0x7FFFu + ((u >> 16) & 1u);   // round-to-nearest-even
    return (unsigned short)(r >> 16);
}

// ---------------------------------------------------------------------------
// Small conversion / data-movement kernels
// ---------------------------------------------------------------------------
__global__ void k_f32_to_bf16(const float* __restrict__ in,
                              unsigned short* __restrict__ out, int n) {
    int i = blockIdx.x * 256 + threadIdx.x;
    if (i < n) out[i] = f2bf(in[i]);
}

// xs[i][d] = emb[fp[i]][d];  also produce bf16 copy
__global__ void k_gather(const int* __restrict__ fp, const float* __restrict__ emb,
                         float* __restrict__ xs, unsigned short* __restrict__ xs_bf) {
    int idx = blockIdx.x * 256 + threadIdx.x;        // 4096*512 threads
    int i = idx >> 9, d = idx & 511;
    float v = emb[(size_t)fp[i] * 512 + d];
    xs[idx] = v;
    xs_bf[idx] = f2bf(v);
}

// wt[l][k][n] = bf16(W[l][n][k])   (3 x 512 x 512)
__global__ void k_transpose_w(const float* __restrict__ W, unsigned short* __restrict__ wt) {
    int idx = blockIdx.x * 256 + threadIdx.x;        // 3*512*512 threads
    int l = idx >> 18;
    int rem = idx & 262143;
    int k = rem >> 9, n = rem & 511;
    wt[idx] = f2bf(W[(size_t)l * 262144 + (size_t)n * 512 + k]);
}

// ---------------------------------------------------------------------------
// Fragment loaders (ISA 7.12.2 layouts)
//   A (16 x 32, bf16): lane = M row; two 16B chunks at K offsets 8*half and
//   16 + 8*half.
//   B (32 x 16, bf16): lane = N column (lane & 15); K gathered like A (8*half, 16 + 8*half).
// ---------------------------------------------------------------------------
__device__ __forceinline__ FragBF load_a(const unsigned short* rowA, int kb, int half) {
    FragBF f;
    f.q[0] = *(const uint4*)(rowA + kb + 8 * half);
    f.q[1] = *(const uint4*)(rowA + kb + 16 + 8 * half);
    return f;
}
__device__ __forceinline__ FragBF load_b(const unsigned short* rowB, int ldb, int half) {
    FragBF f;
    unsigned short t[16];
#pragma unroll
    for (int s = 0; s < 8; ++s) {
        t[s]     = rowB[(size_t)(8 * half + s) * ldb];
        t[8 + s] = rowB[(size_t)(16 + 8 * half + s) * ldb];
    }
    __builtin_memcpy(&f, t, 32);
    return f;
}

#define WMMA_BF16(A_, B_, C_) \
    __builtin_amdgcn_wmma_f32_16x16x32_bf16(false, (A_), false, (B_), (short)0, (C_), false, false)

// ---------------------------------------------------------------------------
// WMMA GEMM #1:  H = relu(A(M x K) * B(K x N) + bias) -> bf16
// 32x32 tile per wave (2x2 register blocking); 8 waves / block.
// ---------------------------------------------------------------------------
__global__ __launch_bounds__(256)
void k_gemm_bias_relu_bf16(const unsigned short* __restrict__ A,
                           const unsigned short* __restrict__ B,
                           const float* __restrict__ bias,
                           unsigned short* __restrict__ H,
                           int Nt, int K, int lda, int ldb, int ldh) {
    const int lane = threadIdx.x & 31;
    const int wave = threadIdx.x >> 5;
    const int tile = blockIdx.x * 8 + wave;       // over (M/32)*(N/32) tiles
    const int tM = tile / Nt;
    const int tN = tile - tM * Nt;
    const int half = lane >> 4;

    const unsigned short* rowA0 = A + (size_t)(tM * 32 + (lane & 15)) * lda;
    const unsigned short* rowA1 = rowA0 + (size_t)16 * lda;

    v8f acc00 = {}, acc01 = {}, acc10 = {}, acc11 = {};
    for (int kb = 0; kb < K; kb += 32) {
        FragBF fa0 = load_a(rowA0, kb, half);
        FragBF fa1 = load_a(rowA1, kb, half);
        const unsigned short* rowB = B + (size_t)kb * ldb + tN * 32 + (lane & 15);
        FragBF fb0 = load_b(rowB, ldb, half);
        FragBF fb1 = load_b(rowB + 16, ldb, half);
        __builtin_prefetch(rowA0 + kb + 128, 0, 1);
        acc00 = WMMA_BF16(fa0.v, fb0.v, acc00);
        acc01 = WMMA_BF16(fa0.v, fb1.v, acc01);
        acc10 = WMMA_BF16(fa1.v, fb0.v, acc10);
        acc11 = WMMA_BF16(fa1.v, fb1.v, acc11);
    }

    const int n0 = tN * 32 + (lane & 15);
    const float bv0 = bias[n0];
    const float bv1 = bias[n0 + 16];
    unsigned short* out0 = H + (size_t)(tM * 32 + half * 8) * ldh + n0;
    unsigned short* out1 = out0 + (size_t)16 * ldh;   // second M sub-tile
#pragma unroll
    for (int r = 0; r < 8; ++r) {
        float v;
        v = acc00[r] + bv0; v = v > 0.0f ? v : 0.0f; out0[(size_t)r * ldh]      = f2bf(v);
        v = acc01[r] + bv1; v = v > 0.0f ? v : 0.0f; out0[(size_t)r * ldh + 16] = f2bf(v);
        v = acc10[r] + bv0; v = v > 0.0f ? v : 0.0f; out1[(size_t)r * ldh]      = f2bf(v);
        v = acc11[r] + bv1; v = v > 0.0f ? v : 0.0f; out1[(size_t)r * ldh + 16] = f2bf(v);
    }
}

// ---------------------------------------------------------------------------
// WMMA GEMM #2:  C(f32, M x N) += A(M x K, bf16) * B(K x N, bf16)  (in place)
// 32x32 tile per wave.
// ---------------------------------------------------------------------------
__global__ __launch_bounds__(256)
void k_gemm_accum_f32(const unsigned short* __restrict__ A,
                      const unsigned short* __restrict__ B,
                      float* __restrict__ C,
                      int Nt, int K, int lda, int ldb, int ldc) {
    const int lane = threadIdx.x & 31;
    const int wave = threadIdx.x >> 5;
    const int tile = blockIdx.x * 8 + wave;
    const int tM = tile / Nt;
    const int tN = tile - tM * Nt;
    const int half = lane >> 4;

    float* c00 = C + (size_t)(tM * 32 + half * 8) * ldc + tN * 32 + (lane & 15);
    float* c10 = c00 + (size_t)16 * ldc;
    v8f acc00, acc01, acc10, acc11;
#pragma unroll
    for (int r = 0; r < 8; ++r) {
        acc00[r] = c00[(size_t)r * ldc];
        acc01[r] = c00[(size_t)r * ldc + 16];
        acc10[r] = c10[(size_t)r * ldc];
        acc11[r] = c10[(size_t)r * ldc + 16];
    }

    const unsigned short* rowA0 = A + (size_t)(tM * 32 + (lane & 15)) * lda;
    const unsigned short* rowA1 = rowA0 + (size_t)16 * lda;
    for (int kb = 0; kb < K; kb += 32) {
        FragBF fa0 = load_a(rowA0, kb, half);
        FragBF fa1 = load_a(rowA1, kb, half);
        const unsigned short* rowB = B + (size_t)kb * ldb + tN * 32 + (lane & 15);
        FragBF fb0 = load_b(rowB, ldb, half);
        FragBF fb1 = load_b(rowB + 16, ldb, half);
        __builtin_prefetch(rowA0 + kb + 128, 0, 1);
        acc00 = WMMA_BF16(fa0.v, fb0.v, acc00);
        acc01 = WMMA_BF16(fa0.v, fb1.v, acc01);
        acc10 = WMMA_BF16(fa1.v, fb0.v, acc10);
        acc11 = WMMA_BF16(fa1.v, fb1.v, acc11);
    }
#pragma unroll
    for (int r = 0; r < 8; ++r) {
        c00[(size_t)r * ldc]      = acc00[r];
        c00[(size_t)r * ldc + 16] = acc01[r];
        c10[(size_t)r * ldc]      = acc10[r];
        c10[(size_t)r * ldc + 16] = acc11[r];
    }
}

// ---------------------------------------------------------------------------
// compound[c] = mean over 4096 rows of xs[:, c]   (512 columns)
// ---------------------------------------------------------------------------
__global__ void k_col_mean(const float* __restrict__ xs, float* __restrict__ out) {
    int c = blockIdx.x * 256 + threadIdx.x;          // 512 threads total
    float acc = 0.0f;
    for (int r = 0; r < 4096; ++r) acc += xs[(size_t)r * 512 + c];
    out[c] = acc * (1.0f / 4096.0f);
}

// ---------------------------------------------------------------------------
// Valid 1D convolution: out[co][x] = b[co] + sum_{ci,dk} in[ci][x+dk]*w[co][ci][dk]
// grid = ( ceil(Lout/256), Cout )
// ---------------------------------------------------------------------------
__global__ void k_conv1d(const float* __restrict__ in, const float* __restrict__ w,
                         const float* __restrict__ b, float* __restrict__ out,
                         int Cin, int K, int Lin, int Lout) {
    int x = blockIdx.x * 256 + threadIdx.x;
    int co = blockIdx.y;
    if (x >= Lout) return;
    float acc = b[co];
    for (int ci = 0; ci < Cin; ++ci) {
        const float* ip = in + (size_t)ci * Lin + x;
        const float* wp = w + (size_t)(co * Cin + ci) * K;
        for (int dk = 0; dk < K; ++dk) acc += ip[dk] * wp[dk];
    }
    out[(size_t)co * Lout + x] = acc;
}

__device__ __forceinline__ float block_reduce(float acc) {
    __shared__ float sm[256];
    sm[threadIdx.x] = acc;
    __syncthreads();
    for (int s = 128; s > 0; s >>= 1) {
        if ((int)threadIdx.x < s) sm[threadIdx.x] += sm[threadIdx.x + s];
        __syncthreads();
    }
    return sm[0];
}

// per-channel mean over L (128 blocks)
__global__ void k_chan_mean(const float* __restrict__ x, int L, float* __restrict__ outMean) {
    int c = blockIdx.x;
    float acc = 0.0f;
    for (int j = threadIdx.x; j < L; j += 256) acc += x[(size_t)c * L + j];
    float s = block_reduce(acc);
    if (threadIdx.x == 0) outMean[c] = s / (float)L;
}

// var[c] = mean_j ( meanArr[(c*L + j) % 128] - x[c][j] )^2   (reproduces reference tile quirk)
__global__ void k_chan_var(const float* __restrict__ x, const float* __restrict__ meanArr,
                           int L, float* __restrict__ outVar) {
    int c = blockIdx.x;
    float acc = 0.0f;
    for (int j = threadIdx.x; j < L; j += 256) {
        float u = meanArr[((unsigned)(c * L + j)) & 127u];
        float d = u - x[(size_t)c * L + j];
        acc += d * d;
    }
    float s = block_reduce(acc);
    if (threadIdx.x == 0) outVar[c] = s / (float)L;
}

// y[row] = (opt relu)( dot(W[row,:], x) + b[row] );  one block per row
__global__ void k_gemv(const float* __restrict__ W, const float* __restrict__ x,
                       const float* __restrict__ b, float* __restrict__ y,
                       int Nin, int doRelu) {
    int row = blockIdx.x;
    float acc = 0.0f;
    for (int k = threadIdx.x; k < Nin; k += 256) acc += W[(size_t)row * Nin + k] * x[k];
    float s = block_reduce(acc);
    if (threadIdx.x == 0) {
        float v = s + b[row];
        y[row] = doRelu ? (v > 0.0f ? v : 0.0f) : v;
    }
}

// ---------------------------------------------------------------------------
// Host-side orchestration
// ---------------------------------------------------------------------------
extern "C" void kernel_launch(void* const* d_in, const int* in_sizes, int n_in,
                              void* d_out, int out_size, void* d_ws, size_t ws_size,
                              hipStream_t stream) {
    (void)in_sizes; (void)n_in; (void)out_size; (void)ws_size;

    // Input order: fingerprints, adjacency, words, emb, gnn_w, gnn_b,
    //              conv_w[0..11], conv_b[0..11], embed_word_w, embed_word_b,
    //              out_w, out_b, inter_w, inter_b
    const int*   fingerprints = (const int*)d_in[0];
    const float* adjacency    = (const float*)d_in[1];
    const float* words        = (const float*)d_in[2];
    const float* emb          = (const float*)d_in[3];
    const float* gnn_w        = (const float*)d_in[4];
    const float* gnn_b        = (const float*)d_in[5];
    const float* conv_w[12];
    const float* conv_b[12];
    for (int i = 0; i < 12; ++i) {
        conv_w[i] = (const float*)d_in[6 + i];
        conv_b[i] = (const float*)d_in[18 + i];
    }
    const float* embed_word_w = (const float*)d_in[30];
    const float* embed_word_b = (const float*)d_in[31];
    const float* out_w        = (const float*)d_in[32];
    const float* out_b        = (const float*)d_in[33];
    const float* inter_w      = (const float*)d_in[34];
    const float* inter_b      = (const float*)d_in[35];
    float* out = (float*)d_out;

    // Workspace carve-out
    size_t off = 0;
    auto alloc = [&](size_t bytes) -> void* {
        void* p = (char*)d_ws + off;
        off += (bytes + 255) & ~(size_t)255;
        return p;
    };
    unsigned short* adj_bf = (unsigned short*)alloc((size_t)4096 * 4096 * 2);
    float*          xs     = (float*)alloc((size_t)4096 * 512 * 4);
    unsigned short* xs_bf  = (unsigned short*)alloc((size_t)4096 * 512 * 2);
    unsigned short* hs_bf  = (unsigned short*)alloc((size_t)4096 * 512 * 2);
    unsigned short* wt_bf  = (unsigned short*)alloc((size_t)3 * 512 * 512 * 2);
    float*          buf1   = (float*)alloc((size_t)28 * 9000 * 4);
    float*          buf2   = (float*)alloc((size_t)64 * 9000 * 4);
    float*          buf3   = (float*)alloc((size_t)128 * 9000 * 4);
    float*          wv     = (float*)alloc(1024 * 4);   // word_vectors
    float*          cat    = (float*)alloc(1024 * 4);   // [compound | protein]
    float*          t1     = (float*)alloc(1024 * 4);
    float*          t2     = (float*)alloc(1024 * 4);

    // --- GNN setup ---
    k_f32_to_bf16<<<(4096 * 4096) / 256, 256, 0, stream>>>(adjacency, adj_bf, 4096 * 4096);
    k_transpose_w<<<(3 * 512 * 512) / 256, 256, 0, stream>>>(gnn_w, wt_bf);
    k_gather<<<(4096 * 512) / 256, 256, 0, stream>>>(fingerprints, emb, xs, xs_bf);

    // --- 3 GNN layers: hs = relu(xs @ W^T + b); xs += adj @ hs ---
    const int Nt = 512 / 32;             // 16 column tiles of 32
    const int tiles = (4096 / 32) * Nt;  // 2048 tiles -> 256 blocks of 8 waves
    for (int l = 0; l < 3; ++l) {
        k_gemm_bias_relu_bf16<<<tiles / 8, 256, 0, stream>>>(
            xs_bf, wt_bf + (size_t)l * 512 * 512, gnn_b + (size_t)l * 512, hs_bf,
            Nt, 512, 512, 512, 512);
        k_gemm_accum_f32<<<tiles / 8, 256, 0, stream>>>(
            adj_bf, hs_bf, xs, Nt, 4096, 4096, 512, 512);
        if (l < 2)
            k_f32_to_bf16<<<(4096 * 512) / 256, 256, 0, stream>>>(xs, xs_bf, 4096 * 512);
    }
    // compound -> cat[0..511]
    k_col_mean<<<2, 256, 0, stream>>>(xs, cat);

    // --- CNN branches ---
    const int KS[4] = {5, 10, 15, 20};
    for (int b = 0; b < 4; ++b) {
        const int k = KS[b];
        const int L1 = 9000 - k + 1;
        const int L2 = L1 - k + 1;
        const int L3 = L2 - k + 1;
        k_conv1d<<<dim3((L1 + 255) / 256, 28), 256, 0, stream>>>(
            words, conv_w[b * 3 + 0], conv_b[b * 3 + 0], buf1, 2, k, 9000, L1);
        k_conv1d<<<dim3((L2 + 255) / 256, 64), 256, 0, stream>>>(
            buf1, conv_w[b * 3 + 1], conv_b[b * 3 + 1], buf2, 28, k, L1, L2);
        k_conv1d<<<dim3((L3 + 255) / 256, 128), 256, 0, stream>>>(
            buf2, conv_w[b * 3 + 2], conv_b[b * 3 + 2], buf3, 64, k, L2, L3);
        k_chan_mean<<<128, 256, 0, stream>>>(buf3, L3, wv + b * 256);
        k_chan_var<<<128, 256, 0, stream>>>(buf3, wv + b * 256, L3, wv + b * 256 + 128);
    }

    // protein = embed_word_w @ wv + embed_word_b -> cat[512..1023]
    k_gemv<<<512, 256, 0, stream>>>(embed_word_w, wv, embed_word_b, cat + 512, 1024, 0);

    // output MLP
    k_gemv<<<1024, 256, 0, stream>>>(out_w, cat, out_b, t1, 1024, 1);
    k_gemv<<<1024, 256, 0, stream>>>(out_w + (size_t)1024 * 1024, t1, out_b + 1024, t2, 1024, 1);
    k_gemv<<<2, 256, 0, stream>>>(inter_w, t2, inter_b, out, 1024, 0);
}
